// GAE_30339648979446
// MI455X (gfx1250) — matrix operand order repaired and checked by reading, not compiled, not run on hardware
//
#include <hip/hip_runtime.h>
#include <hip/hip_fp16.h>

// ---------------------------------------------------------------------------
// GAE forward on MI455X (gfx1250, wave32, WMMA).
// All three GEMMs use v_wmma_f32_16x16x32_f16 (f16 in, f32 accumulate).
// ---------------------------------------------------------------------------

typedef __attribute__((ext_vector_type(16))) _Float16 v16h;
typedef __attribute__((ext_vector_type(8)))  _Float16 v8h;
typedef __attribute__((ext_vector_type(8)))  float    v8f;

#define N_NODES 10000
#define N_EDGES 640000
#define D_IN    512
#define D_HID   256
#define D_OUT   64

static inline unsigned cdiv(long long a, long long b) { return (unsigned)((a + b - 1) / b); }

// ------------------------- small utility kernels ---------------------------

__global__ void k_f32_to_f16(const float* __restrict__ in, _Float16* __restrict__ out, int n) {
    int i = blockIdx.x * blockDim.x + threadIdx.x;
    if (i < n) out[i] = (_Float16)in[i];
}

// in: [K][N] f32 row-major  ->  out: [N][K] f16 (transposed)
__global__ void k_transpose_to_f16(const float* __restrict__ in, _Float16* __restrict__ out,
                                   int K, int N) {
    int i = blockIdx.x * blockDim.x + threadIdx.x;
    if (i < K * N) {
        int k = i / N, n = i - k * N;
        out[(size_t)n * K + k] = (_Float16)in[i];
    }
}

__global__ void k_degrees(const int* __restrict__ src, const int* __restrict__ dst,
                          float* __restrict__ degO, float* __restrict__ degI, int nE) {
    int e = blockIdx.x * blockDim.x + threadIdx.x;
    if (e < nE) {
        atomicAdd(&degO[src[e]], 1.0f);
        atomicAdd(&degI[dst[e]], 1.0f);
    }
}

__global__ void k_rinv(const float* __restrict__ deg, float* __restrict__ rinv, int n) {
    int i = blockIdx.x * blockDim.x + threadIdx.x;
    if (i < n) rinv[i] = rsqrtf(fmaxf(deg[i], 1.0f));
}

// agg[dst[e]*d + c] += h[src[e]*d + c]   (one thread per (edge, column))
__global__ void k_scatter(const float* __restrict__ h, const int* __restrict__ src,
                          const int* __restrict__ dst, float* __restrict__ agg,
                          int ld /*log2(d)*/, long long total) {
    long long i = (long long)blockIdx.x * blockDim.x + threadIdx.x;
    if (i >= total) return;
    int e = (int)(i >> ld);
    int c = (int)(i & ((1 << ld) - 1));
    atomicAdd(&agg[((size_t)dst[e] << ld) + c], h[((size_t)src[e] << ld) + c]);
}

// out = (relu?)(agg * rin[node] + bias[c])  -> f16 for the next WMMA GEMM
__global__ void k_norm_act(const float* __restrict__ agg, const float* __restrict__ rin,
                           const float* __restrict__ bias, _Float16* __restrict__ outH,
                           int ld /*log2(d)*/, int n_total, int do_relu) {
    int i = blockIdx.x * blockDim.x + threadIdx.x;
    if (i >= n_total) return;
    int node = i >> ld;
    int c = i & ((1 << ld) - 1);
    float v = agg[i] * rin[node] + bias[c];
    if (do_relu) v = fmaxf(v, 0.0f);
    outH[i] = (_Float16)v;
}

// ------------------------------ WMMA GEMM ----------------------------------
// C[M][N] = A[M][K] (f16, row-major) x Bt[N][K]^T (f16, stored transposed),
// optionally scaled per output row by rowScale[m]. M,N mult of 16; K mult of 32.
//
// Fragment layouts used here (wave32; contiguous k order, same for A and B):
//   A 16x32:  lane = (half<<4)|m ; v16h elem i -> k = half*16 + i
//             => one contiguous 16xf16 (32B) load from row m.
//   B 32x16:  lane column n = lane&15 ; elem i -> k = half*16 + i
//             => one contiguous 16xf16 (32B) load from Bt row n.
//   C/D f32:  elem r -> row m = half*8 + r, col n = lane&15.
__global__ __launch_bounds__(256)
void k_wmma_gemm(const _Float16* __restrict__ A, const _Float16* __restrict__ Bt,
                 float* __restrict__ C, const float* __restrict__ rowScale,
                 int M, int N, int K) {
    const int wave = threadIdx.x >> 5;
    const int lane = threadIdx.x & 31;
    const int nTN  = N >> 4;
    const int nT   = (M >> 4) * nTN;
    const int tid  = blockIdx.x * 8 + wave;
    if (tid >= nT) return;                 // uniform per wave

    const int tileM = (tid / nTN) << 4;
    const int tileN = (tid % nTN) << 4;
    const int half  = lane >> 4;           // 0 or 1
    const int row16 = lane & 15;

    const _Float16* Arow = A  + (size_t)(tileM + row16) * K + half * 16;
    const _Float16* Brow = Bt + (size_t)(tileN + row16) * K + half * 16;

    v8f acc = {};
    for (int k0 = 0; k0 < K; k0 += 32) {
        v16h a, b;
        a = *(const v16h*)(Arow + k0);                     // k = half*16 + 0..15
        b = *(const v16h*)(Brow + k0);                     // k = half*16 + 0..15
        acc = __builtin_amdgcn_wmma_f32_16x16x32_f16(
                  false, a, false, b, (short)0, acc, false, false);
    }

    float* Cbase = C + (size_t)tileM * N + tileN + row16;
#pragma unroll
    for (int r = 0; r < 8; ++r) {
        int m = half * 8 + r;
        float v = acc[r];
        if (rowScale) v *= rowScale[tileM + m];
        Cbase[(size_t)m * N] = v;
    }
}

// ------------------------------- launcher ----------------------------------

extern "C" void kernel_launch(void* const* d_in, const int* in_sizes, int n_in,
                              void* d_out, int out_size, void* d_ws, size_t ws_size,
                              hipStream_t stream) {
    (void)in_sizes; (void)n_in; (void)out_size; (void)ws_size;

    const float* x   = (const float*)d_in[0];   // [10000, 512]
    const int*   src = (const int*)  d_in[1];   // [640000]
    const int*   dst = (const int*)  d_in[2];   // [640000]
    const float* W1  = (const float*)d_in[3];   // [512, 256]
    const float* b1  = (const float*)d_in[4];   // [256]
    const float* W2  = (const float*)d_in[5];   // [256, 64]
    const float* b2  = (const float*)d_in[6];   // [64]
    float* adj = (float*)d_out;                 // [10000, 10000]

    // ---- workspace layout (256B aligned) ----
    char* ws = (char*)d_ws;
    size_t off = 0;
    auto alloc = [&](size_t bytes) { char* p = ws + off; off += (bytes + 255) & ~(size_t)255; return p; };

    float*    degO  = (float*)   alloc(N_NODES * 4);
    float*    degI  = (float*)   alloc(N_NODES * 4);
    float*    rsO   = (float*)   alloc(N_NODES * 4);
    float*    rsI   = (float*)   alloc(N_NODES * 4);
    _Float16* xh    = (_Float16*)alloc((size_t)N_NODES * D_IN * 2);
    _Float16* w1t   = (_Float16*)alloc((size_t)D_HID * D_IN * 2);
    _Float16* w2t   = (_Float16*)alloc((size_t)D_OUT * D_HID * 2);
    float*    htmp  = (float*)   alloc((size_t)N_NODES * D_HID * 4);  // GEMM1 out (pre-scaled)
    float*    agg1  = (float*)   alloc((size_t)N_NODES * D_HID * 4);
    _Float16* h1h   = (_Float16*)alloc((size_t)N_NODES * D_HID * 2);
    float*    h2tmp = (float*)   alloc((size_t)N_NODES * D_OUT * 4);
    float*    agg2  = (float*)   alloc((size_t)N_NODES * D_OUT * 4);
    _Float16* zh    = (_Float16*)alloc((size_t)N_NODES * D_OUT * 2);

    // ---- zero accumulators (graph-capture-safe) ----
    hipMemsetAsync(degO, 0, N_NODES * 4, stream);
    hipMemsetAsync(degI, 0, N_NODES * 4, stream);
    hipMemsetAsync(agg1, 0, (size_t)N_NODES * D_HID * 4, stream);
    hipMemsetAsync(agg2, 0, (size_t)N_NODES * D_OUT * 4, stream);

    const int T = 256;

    // ---- precision conversions ----
    k_f32_to_f16<<<cdiv((size_t)N_NODES * D_IN, T), T, 0, stream>>>(x, xh, N_NODES * D_IN);
    k_transpose_to_f16<<<cdiv(D_IN * D_HID, T), T, 0, stream>>>(W1, w1t, D_IN, D_HID);
    k_transpose_to_f16<<<cdiv(D_HID * D_OUT, T), T, 0, stream>>>(W2, w2t, D_HID, D_OUT);

    // ---- degrees -> rsqrt(clip(deg,1)) ----
    k_degrees<<<cdiv(N_EDGES, T), T, 0, stream>>>(src, dst, degO, degI, N_EDGES);
    k_rinv<<<cdiv(N_NODES, T), T, 0, stream>>>(degO, rsO, N_NODES);
    k_rinv<<<cdiv(N_NODES, T), T, 0, stream>>>(degI, rsI, N_NODES);

    // ---- layer 1: h = (x @ W1) * rsO ; scatter ; relu(agg * rsI + b1) ----
    {
        int nT = (N_NODES / 16) * (D_HID / 16);
        k_wmma_gemm<<<cdiv(nT, 8), 256, 0, stream>>>(xh, w1t, htmp, rsO, N_NODES, D_HID, D_IN);
    }
    k_scatter<<<cdiv((long long)N_EDGES * D_HID, T), T, 0, stream>>>(
        htmp, src, dst, agg1, 8 /*log2 256*/, (long long)N_EDGES * D_HID);
    k_norm_act<<<cdiv((size_t)N_NODES * D_HID, T), T, 0, stream>>>(
        agg1, rsI, b1, h1h, 8, N_NODES * D_HID, 1);

    // ---- layer 2: h = (h1 @ W2) * rsO ; scatter ; agg * rsI + b2 -> z ----
    {
        int nT = (N_NODES / 16) * (D_OUT / 16);
        k_wmma_gemm<<<cdiv(nT, 8), 256, 0, stream>>>(h1h, w2t, h2tmp, rsO, N_NODES, D_OUT, D_HID);
    }
    k_scatter<<<cdiv((long long)N_EDGES * D_OUT, T), T, 0, stream>>>(
        h2tmp, src, dst, agg2, 6 /*log2 64*/, (long long)N_EDGES * D_OUT);
    k_norm_act<<<cdiv((size_t)N_NODES * D_OUT, T), T, 0, stream>>>(
        agg2, rsI, b2, zh, 6, N_NODES * D_OUT, 0);

    // ---- decoder: adj = z @ z^T  (Bt == z since B = z^T) ----
    {
        int nT = (N_NODES / 16) * (N_NODES / 16);   // 390625 tiles
        k_wmma_gemm<<<cdiv(nT, 8), 256, 0, stream>>>(zh, zh, adj, nullptr,
                                                     N_NODES, N_NODES, D_OUT);
    }
}
